// CPCA_51754355917033
// MI455X (gfx1250) — compile-verified, matrix operand order checked by reading, not run on hardware
//
#include <hip/hip_runtime.h>
#include <hip/hip_bf16.h>

// ---------------------------------------------------------------------------
// CPCA loss on MI455X (gfx1250): f16 WMMA 16x16x32 with f32 accumulation.
// Fused MLP now runs 4 waves per 32-row tile sharing one LDS hidden buffer
// (10 waves/SIMD occupancy during the dominant negatives phase).
// ---------------------------------------------------------------------------

#define N_    128
#define T_    512
#define H_    512
#define K_    16
#define S_    16
#define F_    4
#define EMB_  32
#define NEG_  20
#define B_    (N_ * S_)      // 2048
#define G3H_  (3 * H_)       // 1536
#define L_    (T_ - 1)       // 511

typedef __attribute__((ext_vector_type(16))) _Float16 v16h;
typedef __attribute__((ext_vector_type(8)))  _Float16 v8h;
typedef __attribute__((ext_vector_type(8)))  float    v8f;

// ---- WMMA fragment helpers -------------------------------------------------
// 16-bit A/B fragment layout (wave32): lane L in [0,16) holds row (L) elements
// K = [0..7] and [16..23]; lane L+16 holds row (L) elements K = [8..15] and
// [24..31].  Caller passes pointer to (row-for-this-lane) + ksel, where
// ksel = (lane&16) ? 8 : 0.
__device__ __forceinline__ v16h ld16(const _Float16* p) {
  v8h lo = *(const v8h*)(p);
  v8h hi = *(const v8h*)(p + 16);
  v16h f;
#pragma unroll
  for (int i = 0; i < 8; ++i) { f[i] = lo[i]; f[i + 8] = hi[i]; }
  return f;
}

__device__ __forceinline__ v8f wmma_f16(v16h a, v16h b, v8f c) {
  return __builtin_amdgcn_wmma_f32_16x16x32_f16(
      /*neg_a=*/false, a, /*neg_b=*/false, b,
      /*c_mod=*/(short)0, c, /*reuse_a=*/false, /*reuse_b=*/false);
}

__device__ __forceinline__ float softplus_f(float x) {
  return x > 0.f ? x + log1pf(expf(-x)) : log1pf(expf(x));
}

// ---- generic NT GEMM: C[M,N] (f32) = A[M,K] (f16) @ W[N,K]^T (f16) + bias --
// One wave computes a 32x64 tile; M%32==0, N%64==0, K%32==0 required.
__global__ __launch_bounds__(256)
void gemm_nt_kernel(const _Float16* __restrict__ A, int lda,
                    const _Float16* __restrict__ W, int ldw,
                    const float* __restrict__ bias,
                    float* __restrict__ C, int ldc,
                    int M, int N, int K) {
  int gid  = blockIdx.x * blockDim.x + threadIdx.x;
  int wave = gid >> 5;
  int lane = threadIdx.x & 31;
  int tn_cnt = N >> 6;
  int total  = (M >> 5) * tn_cnt;
  if (wave >= total) return;
  int tm = wave / tn_cnt, tn = wave % tn_cnt;
  int m0 = tm << 5, n0 = tn << 6;
  int mr   = lane & 15;
  int ksel = (lane & 16) ? 8 : 0;

  const _Float16* a0 = A + (size_t)(m0 + mr) * lda + ksel;
  const _Float16* a1 = a0 + (size_t)16 * lda;
  const _Float16* w0 = W + (size_t)(n0 + mr) * ldw + ksel;
  const _Float16* w1 = w0 + (size_t)16 * ldw;
  const _Float16* w2 = w0 + (size_t)32 * ldw;
  const _Float16* w3p = w0 + (size_t)48 * ldw;

  v8f acc[2][4];
#pragma unroll
  for (int i = 0; i < 2; ++i)
#pragma unroll
    for (int j = 0; j < 4; ++j)
#pragma unroll
      for (int v = 0; v < 8; ++v) acc[i][j][v] = 0.f;

  for (int k = 0; k < K; k += 32) {
    v16h af0 = ld16(a0 + k);
    v16h af1 = ld16(a1 + k);
    v16h wf0 = ld16(w0 + k);
    v16h wf1 = ld16(w1 + k);
    v16h wf2 = ld16(w2 + k);
    v16h wf3 = ld16(w3p + k);
    acc[0][0] = wmma_f16(af0, wf0, acc[0][0]);
    acc[0][1] = wmma_f16(af0, wf1, acc[0][1]);
    acc[0][2] = wmma_f16(af0, wf2, acc[0][2]);
    acc[0][3] = wmma_f16(af0, wf3, acc[0][3]);
    acc[1][0] = wmma_f16(af1, wf0, acc[1][0]);
    acc[1][1] = wmma_f16(af1, wf1, acc[1][1]);
    acc[1][2] = wmma_f16(af1, wf2, acc[1][2]);
    acc[1][3] = wmma_f16(af1, wf3, acc[1][3]);
  }

  int r0 = (lane >> 4) << 3;   // C/D layout: row = vgpr + 8*(lane>=16)
  int cc = lane & 15;          //             col = lane & 15
#pragma unroll
  for (int mi = 0; mi < 2; ++mi)
#pragma unroll
    for (int ni = 0; ni < 4; ++ni) {
      int col  = n0 + ni * 16 + cc;
      float bv = bias ? bias[col] : 0.f;
#pragma unroll
      for (int v = 0; v < 8; ++v) {
        int row = m0 + mi * 16 + r0 + v;
        C[(size_t)row * ldc + col] = acc[mi][ni][v] + bv;
      }
    }
}

// ---- elementwise prep kernels ---------------------------------------------
__global__ void f32_to_f16_kernel(const float* __restrict__ in,
                                  _Float16* __restrict__ out, int n) {
  int i = blockIdx.x * blockDim.x + threadIdx.x;
  if (i < n) out[i] = (_Float16)in[i];
}

__global__ void zero3_kernel(float* a) {
  if (threadIdx.x < 3) a[threadIdx.x] = 0.f;
}

// forward mask (cumprod over K of unfolded not_dones) at sampled unroll steps.
__global__ void mask_kernel(const float* __restrict__ not_dones,
                            const int* __restrict__ ti,
                            const int* __restrict__ us,
                            float* __restrict__ mf, float* __restrict__ denom) {
  int b = blockIdx.x * blockDim.x + threadIdx.x;
  if (b >= B_) return;
  int n = b / S_, s = b % S_;
  int t0 = ti[s];
  bool cum = true;
  float local = 0.f;
  for (int k = 0; k < K_; ++k) {
    int idx = k + t0;
    bool valid = idx < L_;
    int ic = valid ? idx : (L_ - 1);
    bool nd = not_dones[(size_t)n * T_ + ic] > 0.f;
    cum = cum && nd && valid;
    float fm = cum ? 1.f : 0.f;
#pragma unroll
    for (int f = 0; f < F_; ++f)
      if (us[f] == k) { mf[f * B_ + b] = fm; local += fm; }
  }
  atomicAdd(denom, local);
}

// a_unf[k, n*S+s, :] = valid ? action_embed[actions[n, k+ti[s]]] : 0   (f16)
__global__ void aunf_kernel(const int* __restrict__ actions,
                            const float* __restrict__ embed,
                            const int* __restrict__ ti,
                            _Float16* __restrict__ aunf) {
  int id = blockIdx.x * blockDim.x + threadIdx.x;
  if (id >= K_ * B_) return;
  int k = id / B_, b = id % B_;
  int n = b / S_, s = b % S_;
  int idx = k + ti[s];
  bool valid = idx < L_;
  int ic = valid ? idx : (L_ - 1);
  int a = actions[(size_t)n * T_ + ic];
  _Float16* o = aunf + (size_t)id * EMB_;
  const float* e = embed + (size_t)a * EMB_;
#pragma unroll
  for (int j = 0; j < EMB_; ++j) o[j] = valid ? (_Float16)e[j] : (_Float16)0.f;
}

// h0[n*S+s, :] = rnn_outputs[n, ti[s], :]
__global__ void h0_kernel(const float* __restrict__ rnn_out,
                          const int* __restrict__ ti,
                          float* __restrict__ h, _Float16* __restrict__ h16) {
  int id = blockIdx.x * blockDim.x + threadIdx.x;
  if (id >= B_ * H_) return;
  int b = id / H_, j = id % H_;
  int n = b / S_, s = b % S_;
  float v = rnn_out[((size_t)n * T_ + ti[s]) * H_ + j];
  h[id] = v;
  h16[id] = (_Float16)v;
}

// tg[f*B+b, :] = (us[f]+ti[s] < L) ? rnn_inputs[n, 1+us[f]+ti[s], :] : 0 (f16)
__global__ void tg_kernel(const float* __restrict__ rnn_in,
                          const int* __restrict__ ti,
                          const int* __restrict__ us,
                          _Float16* __restrict__ tg) {
  int id = blockIdx.x * blockDim.x + threadIdx.x;
  if (id >= F_ * B_ * H_) return;
  int j = id % H_;
  int rb = id / H_;
  int f = rb / B_, b = rb % B_;
  int n = b / S_, s = b % S_;
  int idx2 = us[f] + ti[s];
  bool v2 = idx2 < L_;
  int t = v2 ? idx2 : (L_ - 1);
  float val = v2 ? rnn_in[((size_t)n * T_ + 1 + t) * H_ + j] : 0.f;
  tg[id] = (_Float16)val;
}

// GRU gate math: r=sig(ir+hr), z=sig(iz+hz), g=tanh(ig + r*hg),
// h' = (1-z)*g + z*h.  gi already includes b_ih, gh includes b_hh (so b_hh_g
// correctly sits inside r*hg).
__global__ __launch_bounds__(256)
void gru_gate_kernel(const float* __restrict__ gi, const float* __restrict__ gh,
                     float* __restrict__ h, _Float16* __restrict__ h16,
                     _Float16* __restrict__ preds,
                     const int* __restrict__ us, int k) {
  int id = blockIdx.x * blockDim.x + threadIdx.x;
  if (id >= B_ * H_) return;
  int b = id / H_, j = id % H_;
  const float* gib = gi + (size_t)b * G3H_;
  const float* ghb = gh + (size_t)b * G3H_;
  float r = 1.f / (1.f + expf(-(gib[j] + ghb[j])));
  float z = 1.f / (1.f + expf(-(gib[j + H_] + ghb[j + H_])));
  float g = tanhf(gib[j + 2 * H_] + r * ghb[j + 2 * H_]);
  float hn = (1.f - z) * g + z * h[id];
  h[id] = hn;
  h16[id] = (_Float16)hn;
#pragma unroll
  for (int f = 0; f < F_; ++f)
    if (us[f] == k) preds[(size_t)f * B_ * H_ + id] = (_Float16)hn;
}

// ---- fused discriminator MLP -----------------------------------------------
// Block = 4 waves (128 threads) per 32-row tile.  Wave w owns column chunks
// {2w, 2w+1} of the 8x64 output columns in BOTH layers, sharing one 32 KB
// LDS hidden buffer.  layer1 -> ReLU -> LDS -> layer2 -> ReLU -> dot(w3)
// partials combined via ds_add_f32 -> masked softplus -> one global atomic.
__global__ __launch_bounds__(128)
void mlp_fused_kernel(const _Float16* __restrict__ Abase,   // rows, stride H_
                      const int* __restrict__ ridx,         // nullable gather
                      const float* __restrict__ predA,      // (rows/div) x H_
                      int divisor,                          // NEG_ or 1
                      const float* __restrict__ mf,         // (rows/div)
                      const _Float16* __restrict__ W1b, int ldw1,
                      const float* __restrict__ b1,
                      const _Float16* __restrict__ W2h,
                      const float* __restrict__ b2,
                      const float* __restrict__ w3,
                      const float* __restrict__ b3,
                      float sign,
                      float* __restrict__ loss_accum) {
  __shared__ _Float16 hid[32 * H_];   // 32 KB
  __shared__ float rowacc[32];
  int lane = threadIdx.x & 31;
  int wv   = threadIdx.x >> 5;        // 0..3: column-chunk owner
  int m0 = blockIdx.x * 32;
  int mr = lane & 15;
  int ksel = (lane & 16) ? 8 : 0;
  int r0 = (lane >> 4) << 3;
  int cc = lane & 15;

  if (threadIdx.x < 32) rowacc[threadIdx.x] = 0.f;

  const _Float16* arow[2];
#pragma unroll
  for (int mi = 0; mi < 2; ++mi) {
    int r = m0 + mi * 16 + mr;
    int rr = ridx ? ridx[r] : r;
    arow[mi] = Abase + (size_t)rr * H_ + ksel;
  }

  // ---- layer 1: hid = relu(predA + A @ W1b^T + b1) (this wave's 2 chunks) --
#pragma unroll
  for (int ncq = 0; ncq < 2; ++ncq) {
    int nc = wv * 2 + ncq;
    int n0 = nc * 64;
    const _Float16* w0 = W1b + (size_t)(n0 + mr) * ldw1 + ksel;
    v8f acc[2][4];
#pragma unroll
    for (int i = 0; i < 2; ++i)
#pragma unroll
      for (int j = 0; j < 4; ++j)
#pragma unroll
        for (int v = 0; v < 8; ++v) acc[i][j][v] = 0.f;
    for (int k = 0; k < H_; k += 32) {
      v16h af0 = ld16(arow[0] + k);
      v16h af1 = ld16(arow[1] + k);
      v16h wf0 = ld16(w0 + k);
      v16h wf1 = ld16(w0 + 16 * ldw1 + k);
      v16h wf2 = ld16(w0 + 32 * ldw1 + k);
      v16h wf3 = ld16(w0 + 48 * ldw1 + k);
      acc[0][0] = wmma_f16(af0, wf0, acc[0][0]);
      acc[0][1] = wmma_f16(af0, wf1, acc[0][1]);
      acc[0][2] = wmma_f16(af0, wf2, acc[0][2]);
      acc[0][3] = wmma_f16(af0, wf3, acc[0][3]);
      acc[1][0] = wmma_f16(af1, wf0, acc[1][0]);
      acc[1][1] = wmma_f16(af1, wf1, acc[1][1]);
      acc[1][2] = wmma_f16(af1, wf2, acc[1][2]);
      acc[1][3] = wmma_f16(af1, wf3, acc[1][3]);
    }
#pragma unroll
    for (int mi = 0; mi < 2; ++mi)
#pragma unroll
      for (int ni = 0; ni < 4; ++ni) {
        int col = n0 + ni * 16 + cc;
#pragma unroll
        for (int v = 0; v < 8; ++v) {
          int row = mi * 16 + r0 + v;
          int grow = m0 + row;
          float val = acc[mi][ni][v] +
                      predA[(size_t)(grow / divisor) * H_ + col] + b1[col];
          hid[row * H_ + col] = (_Float16)fmaxf(val, 0.f);
        }
      }
  }
  __syncthreads();

  // ---- layer 2 + 3: rowsum += sum_col relu(hid @ W2^T + b2) * w3 ----
  float rowsum[2][8];
#pragma unroll
  for (int mi = 0; mi < 2; ++mi)
#pragma unroll
    for (int v = 0; v < 8; ++v) rowsum[mi][v] = 0.f;

  const _Float16* h0p = hid + (size_t)mr * H_ + ksel;
#pragma unroll
  for (int ncq = 0; ncq < 2; ++ncq) {
    int nc = wv * 2 + ncq;
    int n0 = nc * 64;
    const _Float16* w0 = W2h + (size_t)(n0 + mr) * H_ + ksel;
    v8f acc[2][4];
#pragma unroll
    for (int i = 0; i < 2; ++i)
#pragma unroll
      for (int j = 0; j < 4; ++j)
#pragma unroll
        for (int v = 0; v < 8; ++v) acc[i][j][v] = 0.f;
    for (int k = 0; k < H_; k += 32) {
      v16h af0 = ld16(h0p + k);
      v16h af1 = ld16(h0p + 16 * H_ + k);
      v16h wf0 = ld16(w0 + k);
      v16h wf1 = ld16(w0 + 16 * H_ + k);
      v16h wf2 = ld16(w0 + 32 * H_ + k);
      v16h wf3 = ld16(w0 + 48 * H_ + k);
      acc[0][0] = wmma_f16(af0, wf0, acc[0][0]);
      acc[0][1] = wmma_f16(af0, wf1, acc[0][1]);
      acc[0][2] = wmma_f16(af0, wf2, acc[0][2]);
      acc[0][3] = wmma_f16(af0, wf3, acc[0][3]);
      acc[1][0] = wmma_f16(af1, wf0, acc[1][0]);
      acc[1][1] = wmma_f16(af1, wf1, acc[1][1]);
      acc[1][2] = wmma_f16(af1, wf2, acc[1][2]);
      acc[1][3] = wmma_f16(af1, wf3, acc[1][3]);
    }
#pragma unroll
    for (int mi = 0; mi < 2; ++mi)
#pragma unroll
      for (int ni = 0; ni < 4; ++ni) {
        int col = n0 + ni * 16 + cc;
        float bv = b2[col], wv3 = w3[col];
#pragma unroll
        for (int v = 0; v < 8; ++v)
          rowsum[mi][v] += fmaxf(acc[mi][ni][v] + bv, 0.f) * wv3;
      }
  }

  // reduce the 16 column-lanes of each half-wave (xor<16 stays in-half),
  // then combine the 4 waves' partial row-dots through LDS.
#pragma unroll
  for (int mi = 0; mi < 2; ++mi)
#pragma unroll
    for (int v = 0; v < 8; ++v) {
      float s = rowsum[mi][v];
#pragma unroll
      for (int d = 1; d < 16; d <<= 1) s += __shfl_xor(s, d, 32);
      rowsum[mi][v] = s;
    }
  if (cc == 0) {  // lanes 0 and 16 own disjoint row sets within each wave
#pragma unroll
    for (int mi = 0; mi < 2; ++mi)
#pragma unroll
      for (int v = 0; v < 8; ++v)
        atomicAdd(&rowacc[mi * 16 + r0 + v], rowsum[mi][v]);
  }
  __syncthreads();

  // ---- masked softplus loss, one global atomic per block ----
  if (threadIdx.x < 32) {
    int grow = m0 + threadIdx.x;
    float logit = rowacc[threadIdx.x] + b3[0];
    float l = softplus_f(sign * logit) * mf[grow / divisor];
#pragma unroll
    for (int d = 1; d < 32; d <<= 1) l += __shfl_xor(l, d, 32);
    if (threadIdx.x == 0) atomicAdd(loss_accum, l);
  }
}

__global__ void finalize_kernel(const float* __restrict__ acc,
                                float* __restrict__ out) {
  if (threadIdx.x == 0) {
    float denom = acc[0];
    out[0] = 0.1f * (acc[1] / denom + acc[2] / (denom * (float)NEG_));
  }
}

// ---------------------------------------------------------------------------
extern "C" void kernel_launch(void* const* d_in, const int* in_sizes, int n_in,
                              void* d_out, int out_size, void* d_ws, size_t ws_size,
                              hipStream_t stream) {
  const int*   actions   = (const int*)d_in[0];
  const float* not_dones = (const float*)d_in[1];
  const float* rnn_in    = (const float*)d_in[2];
  const float* rnn_out   = (const float*)d_in[3];
  const int*   ti        = (const int*)d_in[4];
  const int*   us        = (const int*)d_in[5];
  const int*   neg_idx   = (const int*)d_in[6];
  const float* embed     = (const float*)d_in[7];
  const float* W_ih      = (const float*)d_in[8];
  const float* W_hh      = (const float*)d_in[9];
  const float* b_ih      = (const float*)d_in[10];
  const float* b_hh      = (const float*)d_in[11];
  const float* W1        = (const float*)d_in[12];
  const float* b1        = (const float*)d_in[13];
  const float* W2        = (const float*)d_in[14];
  const float* b2        = (const float*)d_in[15];
  const float* W3        = (const float*)d_in[16];
  const float* b3        = (const float*)d_in[17];
  float* out = (float*)d_out;

  // ---- workspace layout (≈138 MB, 256B aligned slots) ----
  char* base = (char*)d_ws;
  size_t off = 0;
  auto alloc = [&](size_t bytes) {
    char* p = base + off;
    off += (bytes + 255) & ~(size_t)255;
    return p;
  };
  _Float16* rnn16  = (_Float16*)alloc((size_t)N_ * T_ * H_ * 2);
  _Float16* Wih16  = (_Float16*)alloc((size_t)G3H_ * EMB_ * 2);
  _Float16* Whh16  = (_Float16*)alloc((size_t)G3H_ * H_ * 2);
  _Float16* W1h    = (_Float16*)alloc((size_t)H_ * 2 * H_ * 2);
  _Float16* W2h    = (_Float16*)alloc((size_t)H_ * H_ * 2);
  _Float16* aunf   = (_Float16*)alloc((size_t)K_ * B_ * EMB_ * 2);
  float*    h      = (float*)   alloc((size_t)B_ * H_ * 4);
  _Float16* h16    = (_Float16*)alloc((size_t)B_ * H_ * 2);
  float*    gi     = (float*)   alloc((size_t)B_ * G3H_ * 4);
  float*    gh     = (float*)   alloc((size_t)B_ * G3H_ * 4);
  _Float16* preds  = (_Float16*)alloc((size_t)F_ * B_ * H_ * 2);
  float*    predA  = (float*)   alloc((size_t)F_ * B_ * H_ * 4);
  _Float16* tg     = (_Float16*)alloc((size_t)F_ * B_ * H_ * 2);
  float*    mf     = (float*)   alloc((size_t)F_ * B_ * 4);
  float*    accum  = (float*)   alloc(4 * 4);   // [denom, pos_sum, neg_sum]
  (void)ws_size; (void)n_in; (void)in_sizes; (void)out_size;

  auto blks = [](long n) { return (int)((n + 255) / 256); };
  auto gemm_blocks = [](int M, int N) {
    long waves = (long)(M >> 5) * (N >> 6);
    return (int)((waves * 32 + 255) / 256);
  };

  // ---- prep ----
  zero3_kernel<<<1, 32, 0, stream>>>(accum);
  f32_to_f16_kernel<<<blks((long)N_ * T_ * H_), 256, 0, stream>>>(rnn_in, rnn16, N_ * T_ * H_);
  f32_to_f16_kernel<<<blks(G3H_ * EMB_), 256, 0, stream>>>(W_ih, Wih16, G3H_ * EMB_);
  f32_to_f16_kernel<<<blks(G3H_ * H_),   256, 0, stream>>>(W_hh, Whh16, G3H_ * H_);
  f32_to_f16_kernel<<<blks(H_ * 2 * H_), 256, 0, stream>>>(W1, W1h, H_ * 2 * H_);
  f32_to_f16_kernel<<<blks(H_ * H_),     256, 0, stream>>>(W2, W2h, H_ * H_);
  mask_kernel<<<blks(B_), 256, 0, stream>>>(not_dones, ti, us, mf, &accum[0]);
  aunf_kernel<<<blks(K_ * B_), 256, 0, stream>>>(actions, embed, ti, aunf);
  h0_kernel<<<blks((long)B_ * H_), 256, 0, stream>>>(rnn_out, ti, h, h16);
  tg_kernel<<<blks((long)F_ * B_ * H_), 256, 0, stream>>>(rnn_in, ti, us, tg);

  // ---- GRU: K_ sequential steps ----
  for (int k = 0; k < K_; ++k) {
    // gi = x_k @ W_ih^T + b_ih   (K=32)
    gemm_nt_kernel<<<gemm_blocks(B_, G3H_), 256, 0, stream>>>(
        aunf + (size_t)k * B_ * EMB_, EMB_, Wih16, EMB_, b_ih, gi, G3H_,
        B_, G3H_, EMB_);
    // gh = h @ W_hh^T + b_hh     (K=512)
    gemm_nt_kernel<<<gemm_blocks(B_, G3H_), 256, 0, stream>>>(
        h16, H_, Whh16, H_, b_hh, gh, G3H_, B_, G3H_, H_);
    gru_gate_kernel<<<blks((long)B_ * H_), 256, 0, stream>>>(
        gi, gh, h, h16, preds, us, k);
  }

  // ---- predA = preds @ W1[:, :H]^T  (shared across pos + all negatives) ----
  gemm_nt_kernel<<<gemm_blocks(F_ * B_, H_), 256, 0, stream>>>(
      preds, H_, W1h, 2 * H_, nullptr, predA, H_, F_ * B_, H_, H_);

  // ---- positives: softplus(-logit) ----
  mlp_fused_kernel<<<(F_ * B_) / 32, 128, 0, stream>>>(
      tg, nullptr, predA, 1, mf, W1h + H_, 2 * H_, b1, W2h, b2, W3, b3,
      -1.f, &accum[1]);

  // ---- negatives: softplus(+logit), rows gathered via neg_idx ----
  mlp_fused_kernel<<<(F_ * B_ * NEG_) / 32, 128, 0, stream>>>(
      rnn16, neg_idx, predA, NEG_, mf, W1h + H_, 2 * H_, b1, W2h, b2, W3, b3,
      +1.f, &accum[2]);

  finalize_kernel<<<1, 32, 0, stream>>>(accum, out);
}
